// AnemoneSparseMoeBlock_85718957293663
// MI455X (gfx1250) — matrix operand order repaired and checked by reading, not compiled, not run on hardware
//
#include <hip/hip_runtime.h>
#include <hip/hip_bf16.h>
#include <stdint.h>

// Problem constants (from reference): B=4, S=2048 -> T=8192, D=1024, F=4096, E=8, top-2.
#define T_TOK 8192
#define DDIM  1024
#define FDIM  4096
#define NEXP  8

typedef __attribute__((ext_vector_type(16))) __bf16 v16bf;
typedef __attribute__((ext_vector_type(8)))  float  v8f;

// --- fp32 -> bf16 (round-to-nearest-even) helpers -------------------------
__device__ __forceinline__ uint16_t f2bf(float f) {
  uint32_t u; __builtin_memcpy(&u, &f, 4);
  uint32_t r = u + 0x7FFFu + ((u >> 16) & 1u);
  return (uint16_t)(r >> 16);
}
__device__ __forceinline__ uint32_t pk2(float lo, float hi) {
  return (uint32_t)f2bf(lo) | ((uint32_t)f2bf(hi) << 16);
}

// --- async global->LDS copy (CDNA5 GLOBAL_LOAD_ASYNC_TO_LDS_B128) ---------
#if __has_builtin(__builtin_amdgcn_global_load_async_to_lds_b128)
#define HAS_ASYNC_LDS 1
#else
#define HAS_ASYNC_LDS 0
#endif

typedef int v4i __attribute__((vector_size(16)));
typedef __attribute__((address_space(1))) v4i* gas_v4i_ptr;  // global (device) AS
typedef __attribute__((address_space(3))) v4i* las_v4i_ptr;  // LDS AS

__device__ __forceinline__ void cp16_async(void* lds, const void* g) {
#if HAS_ASYNC_LDS
  __builtin_amdgcn_global_load_async_to_lds_b128((gas_v4i_ptr)g, (las_v4i_ptr)lds, 0, 0);
#else
  *(uint4*)lds = *(const uint4*)g;
#endif
}

template <int N>
__device__ __forceinline__ void wait_async() {
#if __has_builtin(__builtin_amdgcn_s_wait_asynccnt)
  __builtin_amdgcn_s_wait_asynccnt(N);
#else
  asm volatile("s_wait_asynccnt %0" ::"i"(N) : "memory");
#endif
}

// --- 1) convert fp32 -> packed bf16 (used for x and all three weights) ----
__global__ void cvt_bf16_kernel(const float* __restrict__ src, uint32_t* __restrict__ dst) {
  size_t j = (size_t)blockIdx.x * blockDim.x + threadIdx.x;  // 4 floats each
  float4 f = ((const float4*)src)[j];
  dst[2 * j]     = pk2(f.x, f.y);
  dst[2 * j + 1] = pk2(f.z, f.w);
}

// --- 2) router: logits, softmax, top-2, scatter into expert lists ---------
__global__ __launch_bounds__(256) void router_kernel(
    const float* __restrict__ x, const float* __restrict__ rw,
    float* __restrict__ logits, int* __restrict__ counts,
    int* __restrict__ list, float* __restrict__ wlist) {
  __shared__ float srw[NEXP * DDIM];  // 32 KB
  const int tid = threadIdx.x;
  for (int i = tid; i < NEXP * DDIM; i += 256) srw[i] = rw[i];
  __syncthreads();

  const int wave = tid >> 5, lane = tid & 31;
  const int t = blockIdx.x * 8 + wave;
  const float* xt = x + (size_t)t * DDIM;

  float acc[NEXP];
#pragma unroll
  for (int e = 0; e < NEXP; ++e) acc[e] = 0.f;
  for (int j = lane; j < DDIM; j += 32) {
    float xv = xt[j];
#pragma unroll
    for (int e = 0; e < NEXP; ++e) acc[e] += xv * srw[e * DDIM + j];
  }
#pragma unroll
  for (int off = 16; off >= 1; off >>= 1) {
#pragma unroll
    for (int e = 0; e < NEXP; ++e) acc[e] += __shfl_xor(acc[e], off, 32);
  }
  if (lane == 0) {
    float mx = acc[0];
#pragma unroll
    for (int e = 1; e < NEXP; ++e) mx = fmaxf(mx, acc[e]);
    float p[NEXP], s = 0.f;
#pragma unroll
    for (int e = 0; e < NEXP; ++e) { p[e] = __expf(acc[e] - mx); s += p[e]; }
    int i0 = 0;
#pragma unroll
    for (int e = 1; e < NEXP; ++e) if (p[e] > p[i0]) i0 = e;
    int i1 = (i0 == 0) ? 1 : 0;
#pragma unroll
    for (int e = 0; e < NEXP; ++e) if (e != i0 && p[e] > p[i1]) i1 = e;
    float inv = 1.f / s;
#pragma unroll
    for (int e = 0; e < NEXP; ++e) logits[(size_t)t * NEXP + e] = acc[e];
    int pos0 = atomicAdd(&counts[i0], 1);
    list[i0 * T_TOK + pos0] = t; wlist[i0 * T_TOK + pos0] = p[i0] * inv;
    int pos1 = atomicAdd(&counts[i1], 1);
    list[i1 * T_TOK + pos1] = t; wlist[i1 * T_TOK + pos1] = p[i1] * inv;
  }
}

// --- 3) exclusive prefix sum over 8 counts --------------------------------
__global__ void offsets_kernel(const int* __restrict__ counts, int* __restrict__ offs) {
  if (threadIdx.x == 0 && blockIdx.x == 0) {
    int run = 0;
    for (int e = 0; e < NEXP; ++e) { offs[e] = run; run += counts[e]; }
  }
}

// --- WMMA fragment loaders (ISA 16-bit A / B layouts, wave32) -------------
// LDS tiles are row-major [row][64 uint32] (= 128 bf16 K values per row).
union AB16 { v16bf v; uint4 q[2]; };

__device__ __forceinline__ void load_a_frag(AB16& f, const uint32_t* lds, int mrow,
                                            int kk, int half) {
  const int base = mrow * 64 + kk * 16 + half * 4;
  f.q[0] = *(const uint4*)&lds[base];
  f.q[1] = *(const uint4*)&lds[base + 8];
}
__device__ __forceinline__ void load_b_frag(AB16& f, const uint32_t* lds, int nrow,
                                            int kk, int half) {
  const int base = nrow * 64 + kk * 16 + half * 4;
  f.q[0] = *(const uint4*)&lds[base];
  f.q[1] = *(const uint4*)&lds[base + 8];
}

// --- 4) grouped GEMM: act = silu(x@gateT) * (x@upT), per expert ------------
// Block tile: 128 tokens x 64 features, K-stage 128, double-buffered async LDS.
__global__ __launch_bounds__(256) void moe_gateup_kernel(
    const uint32_t* __restrict__ xb, const uint32_t* __restrict__ gwb,
    const uint32_t* __restrict__ uwb, const int* __restrict__ counts,
    const int* __restrict__ offs, const int* __restrict__ list,
    uint16_t* __restrict__ act) {
  const int e   = blockIdx.z;
  const int cnt = counts[e];
  const int m0  = blockIdx.y * 128;
  if (m0 >= cnt) return;
  const int f0   = blockIdx.x * 64;
  const int aoff = offs[e];

  __shared__ uint32_t lA[2][128 * 64];   // 64 KB
  __shared__ uint32_t lBg[2][64 * 64];   // 32 KB
  __shared__ uint32_t lBu[2][64 * 64];   // 32 KB
  __shared__ int ltok[128];

  const int tid = threadIdx.x;
  if (tid < 128) {
    int gm = m0 + tid;
    ltok[tid] = list[e * T_TOK + min(gm, cnt - 1)];
  }
  __syncthreads();

  const int arow_ld = tid >> 1, ahalf = tid & 1;  // 2 threads/row, 32 uints each
  const size_t tA = (size_t)ltok[arow_ld];
  const uint32_t* aS = xb + tA * (DDIM / 2) + ahalf * 32;
  const int brow = tid >> 2, bq = tid & 3;        // 4 threads/row, 16 uints each
  const uint32_t* gS = gwb + ((size_t)e * FDIM + f0 + brow) * (DDIM / 2) + bq * 16;
  const uint32_t* uS = uwb + ((size_t)e * FDIM + f0 + brow) * (DDIM / 2) + bq * 16;

  const int wave = tid >> 5, lane = tid & 31;
  const int wm = (wave & 3) * 32, wn = (wave >> 2) * 32;
  const int half = lane >> 4, l16 = lane & 15;

  auto stage = [&](int kb, int buf) {
    const int k0u = kb * 64;  // 128 bf16 = 64 uints per K-stage
#pragma unroll
    for (int c = 0; c < 8; ++c)
      cp16_async(&lA[buf][arow_ld * 64 + ahalf * 32 + c * 4], aS + k0u + c * 4);
#pragma unroll
    for (int c = 0; c < 4; ++c) {
      cp16_async(&lBg[buf][brow * 64 + bq * 16 + c * 4], gS + k0u + c * 4);
      cp16_async(&lBu[buf][brow * 64 + bq * 16 + c * 4], uS + k0u + c * 4);
    }
  };

  v8f cg[2][2] = {}; v8f cu[2][2] = {};

  constexpr int NK = DDIM / 128;  // 8 K-stages
  stage(0, 0);
  for (int kb = 0; kb < NK; ++kb) {
    const int buf = kb & 1;
    __syncthreads();  // everyone done computing on buf^1 before we overwrite it
    if (kb + 1 < NK) { stage(kb + 1, buf ^ 1); wait_async<16>(); }
    else             { wait_async<0>(); }
    __syncthreads();  // buf fully resident in LDS for all waves

#pragma unroll
    for (int kk = 0; kk < 4; ++kk) {
      AB16 a[2], bg[2], bu[2];
#pragma unroll
      for (int fm = 0; fm < 2; ++fm) load_a_frag(a[fm], lA[buf], wm + fm * 16 + l16, kk, half);
#pragma unroll
      for (int fn = 0; fn < 2; ++fn) {
        load_b_frag(bg[fn], lBg[buf], wn + fn * 16 + l16, kk, half);
        load_b_frag(bu[fn], lBu[buf], wn + fn * 16 + l16, kk, half);
      }
#pragma unroll
      for (int fm = 0; fm < 2; ++fm)
#pragma unroll
        for (int fn = 0; fn < 2; ++fn) {
          cg[fm][fn] = __builtin_amdgcn_wmma_f32_16x16x32_bf16(
              false, a[fm].v, false, bg[fn].v, (short)0, cg[fm][fn], false, false);
          cu[fm][fn] = __builtin_amdgcn_wmma_f32_16x16x32_bf16(
              false, a[fm].v, false, bu[fn].v, (short)0, cu[fm][fn], false, false);
        }
    }
  }

  // epilogue: silu(g)*u -> bf16 act rows (compacted by expert offset)
#pragma unroll
  for (int fm = 0; fm < 2; ++fm)
#pragma unroll
    for (int fn = 0; fn < 2; ++fn) {
      const int nloc = wn + fn * 16 + l16;
#pragma unroll
      for (int r = 0; r < 8; ++r) {
        const int mloc = wm + fm * 16 + half * 8 + r;
        const int gm = m0 + mloc;
        if (gm < cnt) {
          float g = cg[fm][fn][r], u = cu[fm][fn][r];
          float a = (g / (1.f + __expf(-g))) * u;
          act[(size_t)(aoff + gm) * FDIM + f0 + nloc] = f2bf(a);
        }
      }
    }
}

// --- 5) grouped GEMM: out += w_t * (act @ down_wT), per expert -------------
__global__ __launch_bounds__(256) void moe_down_kernel(
    const uint32_t* __restrict__ act_u, const uint32_t* __restrict__ dwb,
    const int* __restrict__ counts, const int* __restrict__ offs,
    const int* __restrict__ list, const float* __restrict__ wlist,
    float* __restrict__ out) {
  const int e   = blockIdx.z;
  const int cnt = counts[e];
  const int m0  = blockIdx.y * 128;
  if (m0 >= cnt) return;
  const int d0   = blockIdx.x * 64;
  const int aoff = offs[e];

  __shared__ uint32_t lA[2][128 * 64];  // 64 KB
  __shared__ uint32_t lB[2][64 * 64];   // 32 KB
  __shared__ int   ltok[128];
  __shared__ float lw[128];

  const int tid = threadIdx.x;
  if (tid < 128) {
    int gm = m0 + tid;
    int cl = min(gm, cnt - 1);
    ltok[tid] = list[e * T_TOK + cl];
    lw[tid]   = (gm < cnt) ? wlist[e * T_TOK + gm] : 0.f;
  }
  __syncthreads();

  const int arow_ld = tid >> 1, ahalf = tid & 1;
  const size_t arow = (size_t)aoff + (size_t)min(m0 + arow_ld, cnt - 1);
  const uint32_t* aS = act_u + arow * (FDIM / 2) + ahalf * 32;
  const int brow = tid >> 2, bq = tid & 3;
  const uint32_t* bS = dwb + ((size_t)e * DDIM + d0 + brow) * (FDIM / 2) + bq * 16;

  const int wave = tid >> 5, lane = tid & 31;
  const int wm = (wave & 3) * 32, wn = (wave >> 2) * 32;
  const int half = lane >> 4, l16 = lane & 15;

  auto stage = [&](int kb, int buf) {
    const int k0u = kb * 64;
#pragma unroll
    for (int c = 0; c < 8; ++c)
      cp16_async(&lA[buf][arow_ld * 64 + ahalf * 32 + c * 4], aS + k0u + c * 4);
#pragma unroll
    for (int c = 0; c < 4; ++c)
      cp16_async(&lB[buf][brow * 64 + bq * 16 + c * 4], bS + k0u + c * 4);
  };

  v8f c[2][2] = {};

  constexpr int NK = FDIM / 128;  // 32 K-stages
  stage(0, 0);
  for (int kb = 0; kb < NK; ++kb) {
    const int buf = kb & 1;
    __syncthreads();
    if (kb + 1 < NK) { stage(kb + 1, buf ^ 1); wait_async<12>(); }
    else             { wait_async<0>(); }
    __syncthreads();

#pragma unroll
    for (int kk = 0; kk < 4; ++kk) {
      AB16 a[2], b[2];
#pragma unroll
      for (int fm = 0; fm < 2; ++fm) load_a_frag(a[fm], lA[buf], wm + fm * 16 + l16, kk, half);
#pragma unroll
      for (int fn = 0; fn < 2; ++fn) load_b_frag(b[fn], lB[buf], wn + fn * 16 + l16, kk, half);
#pragma unroll
      for (int fm = 0; fm < 2; ++fm)
#pragma unroll
        for (int fn = 0; fn < 2; ++fn)
          c[fm][fn] = __builtin_amdgcn_wmma_f32_16x16x32_bf16(
              false, a[fm].v, false, b[fn].v, (short)0, c[fm][fn], false, false);
    }
  }

  // epilogue: scale by routing weight, accumulate (exactly 2 adds/element ->
  // commutative fp32 add -> bitwise-deterministic regardless of list order)
#pragma unroll
  for (int fm = 0; fm < 2; ++fm)
#pragma unroll
    for (int fn = 0; fn < 2; ++fn) {
      const int nloc = wn + fn * 16 + l16;
#pragma unroll
      for (int r = 0; r < 8; ++r) {
        const int mloc = wm + fm * 16 + half * 8 + r;
        const int gm = m0 + mloc;
        if (gm < cnt) {
          int t = ltok[mloc];
          float v = c[fm][fn][r] * lw[mloc];
          unsafeAtomicAdd(&out[(size_t)t * DDIM + d0 + nloc], v);
        }
      }
    }
}

// --- host glue -------------------------------------------------------------
extern "C" void kernel_launch(void* const* d_in, const int* in_sizes, int n_in,
                              void* d_out, int out_size, void* d_ws, size_t ws_size,
                              hipStream_t stream) {
  const float* x  = (const float*)d_in[0];
  const float* rw = (const float*)d_in[1];
  const float* gw = (const float*)d_in[2];
  const float* uw = (const float*)d_in[3];
  const float* dw = (const float*)d_in[4];

  float* out    = (float*)d_out;
  float* logits = out + (size_t)T_TOK * DDIM;

  const size_t MB = 1024 * 1024;
  char* ws = (char*)d_ws;
  uint32_t* xb    = (uint32_t*)ws;                        // 16 MB (T x D bf16)
  uint32_t* gwb   = (uint32_t*)(ws + 16 * MB);            // 64 MB (E*F*D bf16)
  uint32_t* uwb   = (uint32_t*)(ws + 80 * MB);            // 64 MB
  uint32_t* dwb   = (uint32_t*)(ws + 144 * MB);           // 64 MB (E*D*F bf16)
  uint16_t* act   = (uint16_t*)(ws + 208 * MB);           // 128 MB (2T x F bf16)
  char*     tail  = ws + 336 * MB;
  int*      list  = (int*)tail;                           // E*T ints
  float*    wlist = (float*)(tail + (size_t)NEXP * T_TOK * 4);
  int*    counts  = (int*)(tail + (size_t)2 * NEXP * T_TOK * 4);
  int*      offs  = counts + NEXP;

  (void)hipMemsetAsync(out, 0, (size_t)T_TOK * DDIM * sizeof(float), stream);
  (void)hipMemsetAsync(counts, 0, NEXP * sizeof(int), stream);

  const int WN = NEXP * FDIM * DDIM;  // elements per big weight tensor
  cvt_bf16_kernel<<<(T_TOK * DDIM / 4) / 256, 256, 0, stream>>>(x, xb);
  cvt_bf16_kernel<<<(WN / 4) / 256, 256, 0, stream>>>(gw, gwb);
  cvt_bf16_kernel<<<(WN / 4) / 256, 256, 0, stream>>>(uw, uwb);
  cvt_bf16_kernel<<<(WN / 4) / 256, 256, 0, stream>>>(dw, dwb);

  router_kernel<<<T_TOK / 8, 256, 0, stream>>>(x, rw, logits, counts, list, wlist);
  offsets_kernel<<<1, 32, 0, stream>>>(counts, offs);

  dim3 g1(FDIM / 64, T_TOK / 128, NEXP);
  moe_gateup_kernel<<<g1, 256, 0, stream>>>(xb, gwb, uwb, counts, offs, list, act);

  dim3 g2(DDIM / 64, T_TOK / 128, NEXP);
  moe_down_kernel<<<g2, 256, 0, stream>>>((const uint32_t*)act, dwb, counts, offs,
                                          list, wlist, out);
}
